// Attention_44796508897936
// MI455X (gfx1250) — compile-verified, matrix operand order checked by reading, not run on hardware
//
#include <hip/hip_runtime.h>
#include <hip/hip_bf16.h>

typedef __attribute__((ext_vector_type(16))) _Float16 v16h;
typedef __attribute__((ext_vector_type(8)))  _Float16 v8h;
typedef __attribute__((ext_vector_type(8)))  float    v8f;
typedef __attribute__((ext_vector_type(4)))  int      v4i;

#define SCALE_ATT 0.125f   // 64^-0.5
#define LN_EPS    1e-5f

#if __has_builtin(__builtin_amdgcn_global_load_async_to_lds_b128) && \
    __has_builtin(__builtin_amdgcn_s_wait_asynccnt)
#define HAVE_ASYNC_LDS 1
#else
#define HAVE_ASYNC_LDS 0
#endif

typedef __attribute__((address_space(1))) v4i as1_v4i;   // global int4
typedef __attribute__((address_space(3))) v4i as3_v4i;   // LDS int4

__device__ __forceinline__ v8f wmma_f16(v16h a, v16h b, v8f c) {
  return __builtin_amdgcn_wmma_f32_16x16x32_f16(false, a, false, b, (short)0, c, false, false);
}

// 16-half WMMA fragment from row-major f16 memory: two contiguous 16B runs.
__device__ __forceinline__ v16h load_frag(const _Float16* __restrict__ p) {
  v16h r;
#pragma unroll
  for (int j = 0; j < 8; ++j) { r[j] = p[j]; r[8 + j] = p[16 + j]; }
  return r;
}

// Copy 8192B (one 8-fragment B panel) global -> LDS, 2 x b128 per thread.
__device__ __forceinline__ void stage_b(const _Float16* __restrict__ g,
                                        _Float16* l, int tid) {
  const _Float16* gp = g + tid * 8;
  _Float16* lp = l + tid * 8;
#if HAVE_ASYNC_LDS
  __builtin_amdgcn_global_load_async_to_lds_b128((as1_v4i*)gp, (as3_v4i*)lp, 0, 0);
  __builtin_amdgcn_global_load_async_to_lds_b128((as1_v4i*)(gp + 2048),
                                                 (as3_v4i*)(lp + 2048), 0, 0);
#else
  *(v8h*)lp = *(const v8h*)gp;
  *(v8h*)(lp + 2048) = *(const v8h*)(gp + 2048);
#endif
}

#if HAVE_ASYNC_LDS
#define ASYNC_WAIT(n) __builtin_amdgcn_s_wait_asynccnt(n)
#else
#define ASYNC_WAIT(n)
#endif

// ---------------------------------------------------------------------------
// 1) f32 weight [K,Nn] -> f16 B-fragment-swizzled: Wf[(kt*(Nn/16)+nt)*512+lane*16]
// ---------------------------------------------------------------------------
__global__ void __launch_bounds__(256) swizzle_w_kernel(
    const float* __restrict__ W, _Float16* __restrict__ Wf, int K, int Nn) {
  int t    = blockIdx.x * 256 + threadIdx.x;
  int lane = t & 31;
  int frag = t >> 5;
  int ntiles = Nn >> 4;
  int nt = frag % ntiles;
  int kt = frag / ntiles;
  if (kt >= (K >> 5)) return;
  int half = lane >> 4, l15 = lane & 15;
  int n = nt * 16 + l15;
  _Float16* dst = Wf + (size_t)frag * 512 + lane * 16;
#pragma unroll
  for (int j = 0; j < 8; ++j) {
    int k0 = kt * 32 + half * 8 + j;
    dst[j]     = (_Float16)W[(size_t)k0 * Nn + n];
    dst[8 + j] = (_Float16)W[(size_t)(k0 + 16) * Nn + n];
  }
}

// ---------------------------------------------------------------------------
// 2) LayerNorm: one wave32 per 256-wide row, f16 row-major output.
// ---------------------------------------------------------------------------
__global__ void __launch_bounds__(256) layernorm_kernel(
    const float* __restrict__ x, const float* __restrict__ g,
    const float* __restrict__ b, _Float16* __restrict__ y, int nrows) {
  int wave = threadIdx.x >> 5, lane = threadIdx.x & 31;
  int row = blockIdx.x * 8 + wave;
  if (row >= nrows) return;
  const float* xr = x + (size_t)row * 256;
  float v[8];
  float s = 0.f;
#pragma unroll
  for (int j = 0; j < 8; ++j) { v[j] = xr[lane * 8 + j]; s += v[j]; }
#pragma unroll
  for (int off = 16; off; off >>= 1) s += __shfl_xor(s, off, 32);
  float mu = s * (1.0f / 256.0f);
  float vs = 0.f;
#pragma unroll
  for (int j = 0; j < 8; ++j) { v[j] -= mu; vs += v[j] * v[j]; }
#pragma unroll
  for (int off = 16; off; off >>= 1) vs += __shfl_xor(vs, off, 32);
  float rstd = rsqrtf(vs * (1.0f / 256.0f) + LN_EPS);
  _Float16* yr = y + (size_t)row * 256;
#pragma unroll
  for (int j = 0; j < 8; ++j) {
    int c = lane * 8 + j;
    yr[c] = (_Float16)(v[j] * rstd * g[c] + b[c]);
  }
}

// ---------------------------------------------------------------------------
// 3) GEMM: C[M,Nn] = A[M,K](f16 rm) @ Bf(swizzled) + bias.
//    8 waves (4M x 2N), block tile 128x128; per-wave 32x64 (acc[2][4]).
//    B panel (8 frags = 8KB per k-step) async-staged into LDS, double-
//    buffered; B fragment reads batched so WMMAs issue back-to-back.
// ---------------------------------------------------------------------------
__global__ void __launch_bounds__(256) gemm_kernel(
    const _Float16* __restrict__ A, const _Float16* __restrict__ Bf,
    const float* __restrict__ bias, void* __restrict__ Cout,
    int M, int K, int Nn, int out_f16) {
  __shared__ _Float16 Bsh[2][8 * 512];   // 2 x 8KB
  int tid  = threadIdx.x;
  int lane = tid & 31;
  int wave = tid >> 5;
  int wm = wave & 3, wn = wave >> 2;
  int m0 = blockIdx.y * 128 + wm * 32;
  int n0 = blockIdx.x * 128 + wn * 64;
  int half = lane >> 4, l15 = lane & 15;
  int ktiles = K >> 5;
  size_t bstride = (size_t)(Nn >> 4) * 512;

  const _Float16* Ap0 = A + (size_t)(m0 + l15) * K + half * 8;
  const _Float16* Ap1 = Ap0 + (size_t)16 * K;
  const _Float16* Bp = Bf + (size_t)blockIdx.x * 8 * 512;
  const _Float16* bs_base = &Bsh[0][0] + wn * 4 * 512 + lane * 16;

  v8f acc[2][4];
  v8f vzero = {0.f, 0.f, 0.f, 0.f, 0.f, 0.f, 0.f, 0.f};
#pragma unroll
  for (int mi = 0; mi < 2; ++mi)
#pragma unroll
    for (int i = 0; i < 4; ++i) acc[mi][i] = vzero;

  stage_b(Bp, &Bsh[0][0], tid);                    // prologue: k-step 0
  for (int kt = 0; kt < ktiles - 1; ++kt) {
    v16h a0 = load_frag(Ap0);                      // A in flight early
    v16h a1 = load_frag(Ap1);
    Ap0 += 32; Ap1 += 32;
    stage_b(Bp + bstride, &Bsh[(kt + 1) & 1][0], tid);
    Bp += bstride;
    ASYNC_WAIT(2);                                 // oldest panel resident
    __syncthreads();
    const _Float16* bs = bs_base + (kt & 1) * (8 * 512);
    v16h bfr[4];
#pragma unroll
    for (int nt = 0; nt < 4; ++nt) bfr[nt] = *(const v16h*)(bs + nt * 512);
#pragma unroll
    for (int nt = 0; nt < 4; ++nt) acc[0][nt] = wmma_f16(a0, bfr[nt], acc[0][nt]);
#pragma unroll
    for (int nt = 0; nt < 4; ++nt) acc[1][nt] = wmma_f16(a1, bfr[nt], acc[1][nt]);
    __syncthreads();                               // reads done before restage
  }
  {
    v16h a0 = load_frag(Ap0);
    v16h a1 = load_frag(Ap1);
    ASYNC_WAIT(0);
    __syncthreads();
    const _Float16* bs = bs_base + ((ktiles - 1) & 1) * (8 * 512);
    v16h bfr[4];
#pragma unroll
    for (int nt = 0; nt < 4; ++nt) bfr[nt] = *(const v16h*)(bs + nt * 512);
#pragma unroll
    for (int nt = 0; nt < 4; ++nt) acc[0][nt] = wmma_f16(a0, bfr[nt], acc[0][nt]);
#pragma unroll
    for (int nt = 0; nt < 4; ++nt) acc[1][nt] = wmma_f16(a1, bfr[nt], acc[1][nt]);
  }

#pragma unroll
  for (int mi = 0; mi < 2; ++mi) {
#pragma unroll
    for (int nt = 0; nt < 4; ++nt) {
      int n = n0 + nt * 16 + l15;
      float bv = bias ? bias[n] : 0.f;
#pragma unroll
      for (int g = 0; g < 8; ++g) {
        int row = m0 + mi * 16 + g + half * 8;
        float val = acc[mi][nt][g] + bv;
        if (out_f16)
          ((_Float16*)Cout)[(size_t)row * Nn + n] = (_Float16)val;
        else
          ((float*)Cout)[(size_t)row * Nn + n] = val;
      }
    }
  }
}

// ---------------------------------------------------------------------------
// 4) Repack vp [8192,1024] f16 into B-frag layout over (key32 x dcol16).
// ---------------------------------------------------------------------------
__global__ void __launch_bounds__(256) repack_v_kernel(
    const _Float16* __restrict__ vp, _Float16* __restrict__ vpf) {
  int t    = blockIdx.x * 256 + threadIdx.x;
  int lane = t & 31;
  int frag = t >> 5;
  int nt = frag & 63;
  int kt = frag >> 6;
  if (kt >= 256) return;
  int half = lane >> 4, l15 = lane & 15;
  int n = nt * 16 + l15;
  _Float16* dst = vpf + (size_t)frag * 512 + lane * 16;
#pragma unroll
  for (int j = 0; j < 8; ++j) {
    int k0 = kt * 32 + half * 8 + j;
    dst[j]     = vp[(size_t)k0 * 1024 + n];
    dst[8 + j] = vp[(size_t)(k0 + 16) * 1024 + n];
  }
}

// ---------------------------------------------------------------------------
// 5) Flash attention: block = (batch, 16-query tile), 8 waves; wave w owns
//    d-slice [w*128, w*128+128). Partial S reduced via ds_add_f32 in LDS.
//    All loop addressing is pointer-bump + immediate offsets.
// ---------------------------------------------------------------------------
__global__ void __launch_bounds__(256) flash_attn_kernel(
    const _Float16* __restrict__ qp, const _Float16* __restrict__ kp,
    const _Float16* __restrict__ vpf, _Float16* __restrict__ xout) {
  __shared__ float    Sred[16 * 32];
  __shared__ _Float16 Pbuf[16 * 32];

  int lane = threadIdx.x & 31, wave = threadIdx.x >> 5;
  int half = lane >> 4, l15 = lane & 15;
  int b  = blockIdx.x >> 7;
  int qt = blockIdx.x & 127;
  size_t qrow0 = (size_t)b * 2048 + (size_t)qt * 16;

  // Q A-fragments for this wave's 128-wide d-slice
  v16h qa[4];
  const _Float16* qbase = qp + (qrow0 + l15) * 1024 + wave * 128 + half * 8;
#pragma unroll
  for (int kc = 0; kc < 4; ++kc) qa[kc] = load_frag(qbase + kc * 32);

  // running pointers (advance by one 32-key block per iteration)
  const _Float16* kbase =
      kp + ((size_t)b * 2048 + l15) * 1024 + wave * 128 + half * 8;
  const _Float16* vbase =
      vpf + ((size_t)(b * 64) * 64 + wave * 8) * 512 + (size_t)lane * 16;

  v8f vzero = {0.f, 0.f, 0.f, 0.f, 0.f, 0.f, 0.f, 0.f};
  v8f o[8];
#pragma unroll
  for (int i = 0; i < 8; ++i) o[i] = vzero;
  float m[8], l[8];
#pragma unroll
  for (int g = 0; g < 8; ++g) { m[g] = -1e30f; l[g] = 0.f; }

  for (int kb = 0; kb < 64; ++kb) {
    __syncthreads();
    Sred[threadIdx.x]       = 0.f;
    Sred[256 + threadIdx.x] = 0.f;
    __syncthreads();

    // V fragments for this key block issued early (consumed after softmax)
    v16h vb[8];
#pragma unroll
    for (int nt = 0; nt < 8; ++nt)
      vb[nt] = *(const v16h*)(vbase + nt * 512);
    vbase += 64 * 512;

    // K fragments, batched ahead of the WMMA chains
    v16h kf0[4], kf1[4];
#pragma unroll
    for (int kc = 0; kc < 4; ++kc) kf0[kc] = load_frag(kbase + kc * 32);
#pragma unroll
    for (int kc = 0; kc < 4; ++kc) kf1[kc] = load_frag(kbase + 16384 + kc * 32);
    v8f s0 = vzero, s1 = vzero;
#pragma unroll
    for (int kc = 0; kc < 4; ++kc) s0 = wmma_f16(qa[kc], kf0[kc], s0);
#pragma unroll
    for (int kc = 0; kc < 4; ++kc) s1 = wmma_f16(qa[kc], kf1[kc], s1);
    kbase += 32 * 1024;
    if (kb < 63) __builtin_prefetch(kbase, 0, 1);

    // cross-wave S reduction (ds_add_f32)
#pragma unroll
    for (int g = 0; g < 8; ++g) {
      int r = g + half * 8;
      atomicAdd(&Sred[r * 32 + l15], s0[g]);
      atomicAdd(&Sred[r * 32 + 16 + l15], s1[g]);
    }
    __syncthreads();

    // online softmax; row stats land in C-layout lane mapping
    float p0[8], p1[8];
#pragma unroll
    for (int g = 0; g < 8; ++g) {
      int r = g + half * 8;
      p0[g] = Sred[r * 32 + l15] * SCALE_ATT;
      p1[g] = Sred[r * 32 + 16 + l15] * SCALE_ATT;
      float t = fmaxf(p0[g], p1[g]);
#pragma unroll
      for (int off = 8; off; off >>= 1) t = fmaxf(t, __shfl_xor(t, off, 16));
      float mn = fmaxf(m[g], t);
      float corr = __expf(m[g] - mn);
      m[g] = mn;
      p0[g] = __expf(p0[g] - mn);
      p1[g] = __expf(p1[g] - mn);
      float rs = p0[g] + p1[g];
#pragma unroll
      for (int off = 8; off; off >>= 1) rs += __shfl_xor(rs, off, 16);
      l[g] = l[g] * corr + rs;
#pragma unroll
      for (int nt = 0; nt < 8; ++nt) o[nt][g] *= corr;
    }

    // wave 0 stages P for A-fragment consumption
    if (wave == 0) {
#pragma unroll
      for (int g = 0; g < 8; ++g) {
        int r = g + half * 8;
        Pbuf[r * 32 + l15]      = (_Float16)p0[g];
        Pbuf[r * 32 + 16 + l15] = (_Float16)p1[g];
      }
    }
    __syncthreads();

    v16h pa;
    {
      const _Float16* pr = &Pbuf[l15 * 32 + half * 8];
#pragma unroll
      for (int j = 0; j < 8; ++j) { pa[j] = pr[j]; pa[8 + j] = pr[16 + j]; }
    }

    // P @ V with preloaded fragments
#pragma unroll
    for (int nt = 0; nt < 8; ++nt) o[nt] = wmma_f16(pa, vb[nt], o[nt]);
  }

  // normalize + write x (f16 row-major)
#pragma unroll
  for (int g = 0; g < 8; ++g) {
    float rinv = 1.0f / l[g];
    size_t row = qrow0 + g + half * 8;
#pragma unroll
    for (int nt = 0; nt < 8; ++nt) {
      xout[row * 1024 + wave * 128 + nt * 16 + l15] =
          (_Float16)(o[nt][g] * rinv);
    }
  }
}

// ---------------------------------------------------------------------------
extern "C" void kernel_launch(void* const* d_in, const int* in_sizes, int n_in,
                              void* d_out, int out_size, void* d_ws,
                              size_t ws_size, hipStream_t stream) {
  const float* q    = (const float*)d_in[0];
  const float* k    = (const float*)d_in[1];
  const float* v    = (const float*)d_in[2];
  const float* ln_g = (const float*)d_in[3];
  const float* ln_b = (const float*)d_in[4];
  const float* Wq = (const float*)d_in[5];  const float* bq = (const float*)d_in[6];
  const float* Wk = (const float*)d_in[7];  const float* bk = (const float*)d_in[8];
  const float* Wv = (const float*)d_in[9];  const float* bv = (const float*)d_in[10];
  const float* Wo = (const float*)d_in[11]; const float* bo = (const float*)d_in[12];
  float* out = (float*)d_out;

  _Float16* w    = (_Float16*)d_ws;
  _Float16* Wqf  = w;
  _Float16* Wkf  = Wqf + 262144;
  _Float16* Wvf  = Wkf + 262144;
  _Float16* Wof  = Wvf + 262144;
  _Float16* lnq  = Wof + 262144;
  _Float16* lnk  = lnq + 2097152;
  _Float16* lnv  = lnk + 2097152;
  _Float16* qp16 = lnv + 2097152;
  _Float16* kp16 = qp16 + 8388608;
  _Float16* vp16 = kp16 + 8388608;
  _Float16* vpf  = vp16 + 8388608;
  _Float16* x16  = vpf + 8388608;

  swizzle_w_kernel<<<64, 256, 0, stream>>>(Wq, Wqf, 256, 1024);
  swizzle_w_kernel<<<64, 256, 0, stream>>>(Wk, Wkf, 256, 1024);
  swizzle_w_kernel<<<64, 256, 0, stream>>>(Wv, Wvf, 256, 1024);
  swizzle_w_kernel<<<64, 256, 0, stream>>>(Wo, Wof, 1024, 256);

  layernorm_kernel<<<1024, 256, 0, stream>>>(q, ln_g, ln_b, lnq, 8192);
  layernorm_kernel<<<1024, 256, 0, stream>>>(k, ln_g, ln_b, lnk, 8192);
  layernorm_kernel<<<1024, 256, 0, stream>>>(v, ln_g, ln_b, lnv, 8192);

  dim3 pgrid(1024 / 128, 8192 / 128);
  gemm_kernel<<<pgrid, 256, 0, stream>>>(lnq, Wqf, bq, qp16, 8192, 256, 1024, 1);
  gemm_kernel<<<pgrid, 256, 0, stream>>>(lnk, Wkf, bk, kp16, 8192, 256, 1024, 1);
  gemm_kernel<<<pgrid, 256, 0, stream>>>(lnv, Wvf, bv, vp16, 8192, 256, 1024, 1);

  repack_v_kernel<<<2048, 256, 0, stream>>>(vp16, vpf);

  flash_attn_kernel<<<512, 256, 0, stream>>>(qp16, kp16, vpf, x16);

  dim3 ogrid(256 / 128, 8192 / 128);
  gemm_kernel<<<ogrid, 256, 0, stream>>>(x16, Wof, bo, out, 8192, 1024, 256, 0);
}
